// GCN_Net_jump_81243601371600
// MI455X (gfx1250) — compile-verified, matrix operand order checked by reading, not run on hardware
//
#include <hip/hip_runtime.h>
#include <math.h>

typedef __attribute__((ext_vector_type(16))) _Float16 v16h;
typedef __attribute__((ext_vector_type(8)))  _Float16 h8;
typedef __attribute__((ext_vector_type(2)))  _Float16 h2;
typedef __attribute__((ext_vector_type(8)))  float    v8f;

#define CH 128   // IN_CH == DIM == 128

// ---------------------------------------------------------------------------
// Degree: deg[dst[e]] += 1
// ---------------------------------------------------------------------------
__global__ void deg_kernel(const int* __restrict__ dst, float* __restrict__ deg, int E) {
    int e = blockIdx.x * blockDim.x + threadIdx.x;
    if (e < E) atomicAdd(&deg[dst[e]], 1.0f);
}

// dinv[i] = rsqrt(deg[i] + 1)   (in place over deg buffer)
__global__ void dinv_kernel(float* __restrict__ d, int n) {
    int i = blockIdx.x * blockDim.x + threadIdx.x;
    if (i < n) d[i] = rsqrtf(d[i] + 1.0f);
}

// ---------------------------------------------------------------------------
// One-shot: Wt[n][k] = (f16) W[k][n]  (transposed f16 copy, 128x128)
// ---------------------------------------------------------------------------
__global__ void wcvt_kernel(const float* __restrict__ W, _Float16* __restrict__ Wt) {
    int i = blockIdx.x * blockDim.x + threadIdx.x;
    if (i < CH * CH) {
        int n = i >> 7;
        int k = i & (CH - 1);
        Wt[(size_t)n * CH + k] = (_Float16)W[(size_t)k * CH + n];
    }
}

// ---------------------------------------------------------------------------
// WMMA GEMM:  out[N x 128] = X[N x 128] @ W[128 x 128]   (f16 in, f32 acc)
// 256 threads = 8 waves; block owns a 16-row tile, wave w owns cols [16w,16w+16).
// X tile staged f32->f16 in LDS; B fragments loaded straight from the
// pre-transposed f16 Wt in global (L0/L2 resident, 32 KB) as b128 runs.
// ---------------------------------------------------------------------------
__global__ void __launch_bounds__(256)
gemm_wmma_kernel(const float* __restrict__ X, const _Float16* __restrict__ Wt,
                 float* __restrict__ out) {
    __shared__ _Float16 sX[16][CH];      // 4 KB, row-major [m][k]

    const int rowBase = blockIdx.x * 16;

    // stage X tile: float4 global loads -> packed h2 LDS stores
    for (int i = threadIdx.x; i < 16 * CH / 4; i += 256) {
        int rr = i >> 5;              // 32 float4 per row
        int c4 = (i & 31) * 4;
        const float4 v = *(const float4*)&X[(size_t)(rowBase + rr) * CH + c4];
        h2 p0 = { (_Float16)v.x, (_Float16)v.y };
        h2 p1 = { (_Float16)v.z, (_Float16)v.w };
        *(h2*)&sX[rr][c4]     = p0;
        *(h2*)&sX[rr][c4 + 2] = p1;
    }
    __syncthreads();

    const int lane = threadIdx.x & 31;
    const int wave = threadIdx.x >> 5;      // 0..7 -> column tile
    const int row  = lane & 15;             // M for A, N for B
    const int sel  = lane >> 4;             // lane half
    const int col  = wave * 16 + row;       // this lane's B column

    const _Float16* wrow = Wt + (size_t)col * CH;

    v8f acc = {};
#pragma unroll
    for (int k0 = 0; k0 < CH; k0 += 32) {
        // ISA 16-bit fragment: element j -> K = ((j<8)?j:j+8) + sel*8
        // i.e. two contiguous 8-half (16 B) runs per operand per chunk.
        h8 a0 = *(const h8*)&sX[row][k0 + sel * 8];
        h8 a1 = *(const h8*)&sX[row][k0 + 16 + sel * 8];
        h8 b0 = *(const h8*)&wrow[k0 + sel * 8];
        h8 b1 = *(const h8*)&wrow[k0 + 16 + sel * 8];
        v16h a = __builtin_shufflevector(a0, a1, 0,1,2,3,4,5,6,7,8,9,10,11,12,13,14,15);
        v16h b = __builtin_shufflevector(b0, b1, 0,1,2,3,4,5,6,7,8,9,10,11,12,13,14,15);
        acc = __builtin_amdgcn_wmma_f32_16x16x32_f16(
            /*neg_a=*/false, a, /*neg_b=*/false, b,
            /*c_mod=*/(short)0, acc, /*reuse_a=*/false, /*reuse_b=*/false);
    }

    // C/D layout: VGPR r -> M = r + 8*sel, N = lane%16
#pragma unroll
    for (int r = 0; r < 8; ++r) {
        int m = r + sel * 8;
        out[(size_t)(rowBase + m) * CH + wave * 16 + row] = acc[r];
    }
}

// ---------------------------------------------------------------------------
// Edge aggregation: agg[dst] += h[src] * dinv[src]*dinv[dst]
// 256-thread block covers 2 edges; coalesced row gather + f32 atomic scatter.
// ---------------------------------------------------------------------------
__global__ void __launch_bounds__(256)
edge_agg_kernel(const float* __restrict__ h, const float* __restrict__ dinv,
                const int* __restrict__ src, const int* __restrict__ dst,
                float* __restrict__ agg, int E) {
    int e = blockIdx.x * 2 + (threadIdx.x >> 7);
    if (e >= E) return;
    int c = threadIdx.x & (CH - 1);
    int s = src[e], d = dst[e];
    float nrm = dinv[s] * dinv[d];
    float v = h[(size_t)s * CH + c] * nrm;
    atomicAdd(&agg[(size_t)d * CH + c], v);
}

// ---------------------------------------------------------------------------
// Finalize: v = relu(agg + h*dinv^2 + b); optionally store h_out;
// accumulate mean-pool sums (and graph counts once).
// ---------------------------------------------------------------------------
__global__ void __launch_bounds__(128)
finalize_kernel(const float* __restrict__ h, const float* __restrict__ agg,
                const float* __restrict__ dinv, const float* __restrict__ bias,
                const int* __restrict__ batch, float* __restrict__ h_out,
                float* __restrict__ pool, float* __restrict__ cnt,
                int n, int do_cnt) {
    int node = blockIdx.x;
    if (node >= n) return;
    int c = threadIdx.x;
    float di = dinv[node];
    float v = agg[(size_t)node * CH + c] + h[(size_t)node * CH + c] * di * di + bias[c];
    v = fmaxf(v, 0.0f);
    if (h_out) h_out[(size_t)node * CH + c] = v;
    int g = batch[node];
    atomicAdd(&pool[(size_t)g * CH + c], v);
    if (do_cnt && c == 0) atomicAdd(&cnt[g], 1.0f);
}

// ---------------------------------------------------------------------------
// Head: per graph g: gcat = [pool1/cnt, pool2/cnt]; z = relu(gcat@L1W+L1b);
// out = log_softmax(z@L2W + L2b). One block (128 threads) per graph.
// ---------------------------------------------------------------------------
__global__ void __launch_bounds__(128)
head_kernel(const float* __restrict__ pool1, const float* __restrict__ pool2,
            const float* __restrict__ cnt,
            const float* __restrict__ L1W, const float* __restrict__ L1b,
            const float* __restrict__ L2W, const float* __restrict__ L2b,
            float* __restrict__ out, int G) {
    __shared__ float gfeat[2 * CH];
    __shared__ float s0[CH], s1[CH];
    int g = blockIdx.x;
    if (g >= G) return;
    int t = threadIdx.x;
    float cinv = 1.0f / fmaxf(cnt[g], 1.0f);
    gfeat[t]      = pool1[(size_t)g * CH + t] * cinv;
    gfeat[CH + t] = pool2[(size_t)g * CH + t] * cinv;
    __syncthreads();

    float acc = L1b[t];
#pragma unroll 4
    for (int k = 0; k < 2 * CH; ++k) acc += gfeat[k] * L1W[k * CH + t];
    float z = fmaxf(acc, 0.0f);

    s0[t] = z * L2W[t * 2 + 0];
    s1[t] = z * L2W[t * 2 + 1];
    __syncthreads();
    for (int off = CH / 2; off > 0; off >>= 1) {
        if (t < off) { s0[t] += s0[t + off]; s1[t] += s1[t + off]; }
        __syncthreads();
    }
    if (t == 0) {
        float o0 = s0[0] + L2b[0];
        float o1 = s1[0] + L2b[1];
        float m = fmaxf(o0, o1);
        float lse = m + logf(expf(o0 - m) + expf(o1 - m));
        out[g * 2 + 0] = o0 - lse;
        out[g * 2 + 1] = o1 - lse;
    }
}

// ---------------------------------------------------------------------------
// Launch
// ---------------------------------------------------------------------------
extern "C" void kernel_launch(void* const* d_in, const int* in_sizes, int n_in,
                              void* d_out, int out_size, void* d_ws, size_t ws_size,
                              hipStream_t stream) {
    const float* x    = (const float*)d_in[0];
    const int*   ei   = (const int*)d_in[1];
    const int*   batch= (const int*)d_in[2];
    const float* W1   = (const float*)d_in[3];
    const float* b1   = (const float*)d_in[4];
    const float* W2   = (const float*)d_in[5];
    const float* b2   = (const float*)d_in[6];
    const float* L1W  = (const float*)d_in[7];
    const float* L1b  = (const float*)d_in[8];
    const float* L2W  = (const float*)d_in[9];
    const float* L2b  = (const float*)d_in[10];
    float* out = (float*)d_out;

    const int N = in_sizes[0] / CH;     // 50000
    const int E = in_sizes[1] / 2;      // 800000
    const int G = out_size / 2;         // 256
    const int* src = ei;
    const int* dst = ei + E;

    // workspace layout (floats; every region 16B aligned)
    const size_t NB   = (size_t)N * CH;
    const size_t Nal  = (size_t)((N + 63) / 64) * 64;
    float* ws    = (float*)d_ws;
    float* dinv  = ws;                   // N (deg, then rsqrt in place)
    float* hA    = dinv + Nal;           // N*128  (GEMM output / pre-agg h)
    float* agg   = hA + NB;              // N*128
    float* h1    = agg + NB;             // N*128  (layer-1 activations)
    float* pool1 = h1 + NB;              // G*128
    float* pool2 = pool1 + (size_t)G * CH;
    float* cnt   = pool2 + (size_t)G * CH;        // G (pad to 64)
    _Float16* Wt1 = (_Float16*)(cnt + 64);        // 128*128 halves (32 KB)
    _Float16* Wt2 = Wt1 + (size_t)CH * CH;        // 128*128 halves (32 KB)

    // zero accumulators
    hipMemsetAsync(dinv,  0, Nal * sizeof(float), stream);
    hipMemsetAsync(agg,   0, NB * sizeof(float), stream);
    hipMemsetAsync(pool1, 0, (size_t)G * CH * sizeof(float), stream);
    hipMemsetAsync(pool2, 0, (size_t)G * CH * sizeof(float), stream);
    hipMemsetAsync(cnt,   0, (size_t)G * sizeof(float), stream);

    // one-shot weight conversion (f32 row-major -> f16 transposed)
    wcvt_kernel<<<(CH * CH + 255) / 256, 256, 0, stream>>>(W1, Wt1);
    wcvt_kernel<<<(CH * CH + 255) / 256, 256, 0, stream>>>(W2, Wt2);

    // degrees + normalization
    deg_kernel<<<(E + 255) / 256, 256, 0, stream>>>(dst, dinv, E);
    dinv_kernel<<<(N + 255) / 256, 256, 0, stream>>>(dinv, N);

    const int rowTiles = N / 16;        // 50000 / 16 = 3125 exact
    const int aggBlocks = (E + 1) / 2;

    // ---- layer 1 ----
    gemm_wmma_kernel<<<rowTiles, 256, 0, stream>>>(x, Wt1, hA);
    edge_agg_kernel<<<aggBlocks, 256, 0, stream>>>(hA, dinv, src, dst, agg, E);
    finalize_kernel<<<N, 128, 0, stream>>>(hA, agg, dinv, b1, batch, h1, pool1, cnt, N, 1);

    // ---- layer 2 ----
    hipMemsetAsync(agg, 0, NB * sizeof(float), stream);
    gemm_wmma_kernel<<<rowTiles, 256, 0, stream>>>(h1, Wt2, hA);
    edge_agg_kernel<<<aggBlocks, 256, 0, stream>>>(hA, dinv, src, dst, agg, E);
    finalize_kernel<<<N, 128, 0, stream>>>(hA, agg, dinv, b2, batch, nullptr, pool2, cnt, N, 0);

    // ---- head ----
    head_kernel<<<G, 128, 0, stream>>>(pool1, pool2, cnt, L1W, L1b, L2W, L2b, out, G);
}
